// FluidGPT_2499670966469
// MI455X (gfx1250) — matrix operand order repaired and checked by reading, not compiled, not run on hardware
//
#include <hip/hip_runtime.h>
#include <hip/hip_bf16.h>

// SwinV2 window attention, fused per-window, bf16 WMMA on gfx1250 (wave32).
// Shapes: 4096 windows x 64 tokens x 256 ch, 8 heads x 32 head_dim.

#define NWIN 4096
#define NTOK 64
#define EDIM 256
#define NHEAD 8
#define HDIM 32

// LDS strides (padded for bank-conflict-free fragment loads)
#define XSTR 264   // bf16 row stride for xb/qb/kb/aob
#define VSTR 72    // bf16 stride for transposed V [channel][token]
#define SSTR 68    // f32 stride for scores [64][SSTR]

// LDS layout offsets (bytes)
#define OFF_XB     0
#define OFF_QB     33792            // 64*264*2
#define OFF_KB     67584
#define OFF_VBT    101376
#define OFF_SC     138240           // + 256*72*2
#define OFF_QN     155648           // + 64*68*4
#define OFF_KN     157696
#define SMEM_BYTES 159744

typedef __attribute__((ext_vector_type(16))) __bf16 v16bf;
typedef __attribute__((ext_vector_type(8)))  float  v8f;

// ---- WMMA fragment loaders, per CDNA5 ISA 7.12.2 layouts -------------------
// A (16x32 bf16, MxK): lane&15 = M row; hi=lane>>4; element i holds
// K = (i<8?0:16) + hi*8 + (i&7)  -> two 16B-contiguous runs per lane.
__device__ __forceinline__ v16bf load_a_bf16(const __bf16* p, int row0, int k0, int ld) {
  const int lane = threadIdx.x & 31;
  const int hi   = lane >> 4;
  const __bf16* r = p + (size_t)(row0 + (lane & 15)) * ld + k0 + hi * 8;
  v16bf a;
#pragma unroll
  for (int i = 0; i < 8; ++i) a[i] = r[i];
#pragma unroll
  for (int i = 0; i < 8; ++i) a[8 + i] = r[16 + i];
  return a;
}

// Same A layout but source is f32 (softmax probs), convert on load.
__device__ __forceinline__ v16bf load_a_f32(const float* p, int row0, int k0, int ld) {
  const int lane = threadIdx.x & 31;
  const int hi   = lane >> 4;
  const float* r = p + (size_t)(row0 + (lane & 15)) * ld + k0 + hi * 8;
  v16bf a;
#pragma unroll
  for (int i = 0; i < 8; ++i) a[i] = (__bf16)r[i];
#pragma unroll
  for (int i = 0; i < 8; ++i) a[8 + i] = (__bf16)r[16 + i];
  return a;
}

// B (32x16 bf16, KxN): lane&15 = N col; element i holds K = (i<8?0:16) + hi*8 + (i&7).
// Source stored column-contiguous-in-K: p + n*ldk + k  (so 2 x b128 per lane).
__device__ __forceinline__ v16bf load_b_bf16(const __bf16* p, int k0, int n0, int ldk) {
  const int lane = threadIdx.x & 31;
  const int hi   = lane >> 4;
  const __bf16* c = p + (size_t)(n0 + (lane & 15)) * ldk + k0 + hi * 8;
  v16bf b;
#pragma unroll
  for (int i = 0; i < 8; ++i) b[i] = c[i];
#pragma unroll
  for (int i = 0; i < 8; ++i) b[8 + i] = c[16 + i];
  return b;
}

#define WMMA_BF16(A, B, C) \
  __builtin_amdgcn_wmma_f32_16x16x32_bf16(false, (A), false, (B), (short)0, (C), false, false)

// ---- prologue: CPB MLP table (225 x 8) -------------------------------------
__global__ void cpb_table_kernel(const float* __restrict__ w1, const float* __restrict__ b1,
                                 const float* __restrict__ w2, float* __restrict__ table) {
  __shared__ float hbuf[512];
  const int p = blockIdx.x;              // 0..224
  const int i = p / 15, j = p % 15;
  float th = (float)(i - 7) * (8.0f / 7.0f);
  float tw = (float)(j - 7) * (8.0f / 7.0f);
  float sh = (th > 0.f) ? 1.f : ((th < 0.f) ? -1.f : 0.f);
  float sw = (tw > 0.f) ? 1.f : ((tw < 0.f) ? -1.f : 0.f);
  float c0 = sh * log2f(fabsf(th) + 1.0f) * (1.0f / 3.0f);
  float c1 = sw * log2f(fabsf(tw) + 1.0f) * (1.0f / 3.0f);
  for (int k = threadIdx.x; k < 512; k += blockDim.x)
    hbuf[k] = fmaxf(c0 * w1[k * 2 + 0] + c1 * w1[k * 2 + 1] + b1[k], 0.0f);
  __syncthreads();
  if (threadIdx.x < NHEAD) {
    const int hh = threadIdx.x;
    float s = 0.0f;
    for (int k = 0; k < 512; ++k) s += hbuf[k] * w2[hh * 512 + k];
    table[p * NHEAD + hh] = s;
  }
}

// ---- prologue: weight bf16 conversion + rpb bias (16*sigmoid, gathered) ----
__global__ void prep_convert_kernel(const float* __restrict__ qkvw, const float* __restrict__ projw,
                                    const float* __restrict__ table,
                                    __bf16* __restrict__ qkvw_bf, __bf16* __restrict__ projw_bf,
                                    float* __restrict__ rpb) {
  const int id = blockIdx.x * blockDim.x + threadIdx.x;      // < 196608
  qkvw_bf[id] = (__bf16)qkvw[id];
  if (id < EDIM * EDIM) projw_bf[id] = (__bf16)projw[id];
  if (id < NHEAD * NTOK * NTOK) {
    const int h = id >> 12, i = (id >> 6) & 63, j = id & 63;
    const int d0 = (i >> 3) - (j >> 3) + 7;
    const int d1 = (i & 7) - (j & 7) + 7;
    const float t = table[(d0 * 15 + d1) * NHEAD + h];
    rpb[id] = 16.0f / (1.0f + __expf(-t));
  }
}

// ---- fused per-window attention --------------------------------------------
__global__ void __launch_bounds__(256)
winattn_kernel(const float* __restrict__ x,
               const __bf16* __restrict__ qkvw,     // [768][256] bf16 row-major
               const float* __restrict__ q_bias,
               const float* __restrict__ v_bias,
               const float* __restrict__ flex,
               const float* __restrict__ rpb,       // [8][64][64], incl 16*sigmoid
               const __bf16* __restrict__ projw,    // [256][256] bf16 row-major
               const float* __restrict__ proj_b,
               float* __restrict__ out) {
  extern __shared__ char smem[];
  __bf16* xb     = (__bf16*)(smem + OFF_XB);    // x tile [64][XSTR] (reused as aob)
  __bf16* qb     = (__bf16*)(smem + OFF_QB);    // q bf16 [64][XSTR]
  __bf16* kb     = (__bf16*)(smem + OFF_KB);    // k bf16 [64][XSTR]
  __bf16* vbt    = (__bf16*)(smem + OFF_VBT);   // v bf16 transposed [256][VSTR]
  float*  scores = (float*)(smem + OFF_SC);     // [64][SSTR]
  float*  qnorm  = (float*)(smem + OFF_QN);     // [64*8] 1/||q||
  float*  knorm  = (float*)(smem + OFF_KN);     // [64*8] 1/||k||
  __bf16* aob    = xb;                          // attn output [64][XSTR]

  const int tid  = threadIdx.x;
  const int wave = tid >> 5;
  const int lane = tid & 31;
  const int l16  = lane & 15;
  const int hi   = lane >> 4;
  const int win  = blockIdx.x;

  // Stage 1: x (fp32, global) -> bf16 LDS, b128 loads
  const float4* xw4 = (const float4*)(x + (size_t)win * (NTOK * EDIM));
  for (int e4 = tid; e4 < (NTOK * EDIM) / 4; e4 += 256) {
    const int tok = e4 >> 6;             // 64 float4 per token
    const int c4  = (e4 & 63) * 4;
    float4 t = xw4[e4];
    __bf16* d = xb + tok * XSTR + c4;
    d[0] = (__bf16)t.x; d[1] = (__bf16)t.y; d[2] = (__bf16)t.z; d[3] = (__bf16)t.w;
  }
  __syncthreads();

  // Stage 2: QKV = x @ qkv_w^T + [q_bias,0,v_bias]. 4 Mtiles x 48 Ntiles.
  for (int t = wave; t < 192; t += 8) {
    const int mt = t & 3, nt = t >> 2;
    const int o  = nt * 16 + l16;        // output column 0..767
    const float bv = (o < 256) ? q_bias[o] : ((o < 512) ? 0.0f : v_bias[o - 512]);
    v8f acc = {bv, bv, bv, bv, bv, bv, bv, bv};
#pragma unroll
    for (int kk = 0; kk < 8; ++kk) {
      v16bf a = load_a_bf16(xb, mt * 16, kk * 32, XSTR);
      v16bf b = load_b_bf16(qkvw, kk * 32, nt * 16, EDIM);   // weights via L2
      acc = WMMA_BF16(a, b, acc);
    }
    const int rowb = mt * 16 + hi * 8;
    if (nt < 16) {
#pragma unroll
      for (int r = 0; r < 8; ++r) qb[(rowb + r) * XSTR + o] = (__bf16)acc[r];
    } else if (nt < 32) {
      const int c = o - 256;
#pragma unroll
      for (int r = 0; r < 8; ++r) kb[(rowb + r) * XSTR + c] = (__bf16)acc[r];
    } else {
      const int c = o - 512;
#pragma unroll
      for (int r = 0; r < 8; ++r) vbt[c * VSTR + (rowb + r)] = (__bf16)acc[r];
    }
  }
  __syncthreads();

  // Stage 3: per-(token,head) inverse norms (cosine attention as post-scaling)
  for (int p = tid; p < NTOK * NHEAD; p += 256) {
    const int tok = p >> 3, h = p & 7;
    const __bf16* qp = qb + tok * XSTR + h * HDIM;
    const __bf16* kp = kb + tok * XSTR + h * HDIM;
    float sq = 0.f, sk = 0.f;
#pragma unroll
    for (int c = 0; c < HDIM; ++c) {
      float a = (float)qp[c]; sq += a * a;
      float b = (float)kp[c]; sk += b * b;
    }
    qnorm[p] = 1.0f / fmaxf(sqrtf(sq), 1e-12f);
    knorm[p] = 1.0f / fmaxf(sqrtf(sk), 1e-12f);
  }
  __syncthreads();

  // Stage 4/5: per-head S = scale*(q.k)/(|q||k|) + rpb; softmax; O = P @ V
  for (int h = 0; h < NHEAD; ++h) {
    const float tsc = __expf(fminf(flex[h], 4.6051701860f));  // ln(100)
    // QK^T: 4x4 tiles, 2 per wave, single K=32 WMMA step
    for (int t = wave; t < 16; t += 8) {
      const int mt = t & 3, nt = t >> 2;
      v8f acc = {};
      v16bf a = load_a_bf16(qb, mt * 16, h * HDIM, XSTR);
      v16bf b = load_b_bf16(kb, h * HDIM, nt * 16, XSTR);
      acc = WMMA_BF16(a, b, acc);
      const int rowb = mt * 16 + hi * 8;
      const int j = nt * 16 + l16;
      const float knj = knorm[j * 8 + h] * tsc;
#pragma unroll
      for (int r = 0; r < 8; ++r) {
        const int i = rowb + r;
        scores[i * SSTR + j] = acc[r] * qnorm[i * 8 + h] * knj + rpb[h * 4096 + i * 64 + j];
      }
    }
    __syncthreads();
    // softmax: 8 rows per wave, 2 elems per lane, wave32 shuffle reductions
#pragma unroll 1
    for (int rr = 0; rr < 8; ++rr) {
      const int row = wave * 8 + rr;
      float v0 = scores[row * SSTR + lane];
      float v1 = scores[row * SSTR + 32 + lane];
      float m = fmaxf(v0, v1);
#pragma unroll
      for (int off = 16; off > 0; off >>= 1) m = fmaxf(m, __shfl_xor(m, off, 32));
      float e0 = __expf(v0 - m), e1 = __expf(v1 - m);
      float s = e0 + e1;
#pragma unroll
      for (int off = 16; off > 0; off >>= 1) s += __shfl_xor(s, off, 32);
      const float inv = 1.0f / s;
      scores[row * SSTR + lane]      = e0 * inv;
      scores[row * SSTR + 32 + lane] = e1 * inv;
    }
    __syncthreads();
    // O_head = P(64x64) @ V_head(64x32): 4x2 tiles, exactly 1 per wave
    {
      const int mt = wave & 3, nt = wave >> 2;
      v8f acc = {};
#pragma unroll
      for (int kk = 0; kk < 2; ++kk) {
        v16bf a = load_a_f32(scores, mt * 16, kk * 32, SSTR);
        v16bf b = load_b_bf16(vbt + h * HDIM * VSTR, kk * 32, nt * 16, VSTR);
        acc = WMMA_BF16(a, b, acc);
      }
      const int rowb = mt * 16 + hi * 8;
      const int c = h * HDIM + nt * 16 + l16;
#pragma unroll
      for (int r = 0; r < 8; ++r) aob[(rowb + r) * XSTR + c] = (__bf16)acc[r];
    }
    __syncthreads();
  }

  // Stage 6: out = attn_out @ proj_w^T + proj_b  (4x16 tiles, 8 per wave)
  float* ow = out + (size_t)win * (NTOK * EDIM);
  for (int t = wave; t < 64; t += 8) {
    const int mt = t & 3, nt = t >> 2;
    const int o = nt * 16 + l16;
    const float bv = proj_b[o];
    v8f acc = {bv, bv, bv, bv, bv, bv, bv, bv};
#pragma unroll
    for (int kk = 0; kk < 8; ++kk) {
      v16bf a = load_a_bf16(aob, mt * 16, kk * 32, XSTR);
      v16bf b = load_b_bf16(projw, kk * 32, nt * 16, EDIM);
      acc = WMMA_BF16(a, b, acc);
    }
    const int rowb = mt * 16 + hi * 8;
#pragma unroll
    for (int r = 0; r < 8; ++r) ow[(rowb + r) * EDIM + o] = acc[r];
  }
}

// ---- host entry -------------------------------------------------------------
extern "C" void kernel_launch(void* const* d_in, const int* in_sizes, int n_in,
                              void* d_out, int out_size, void* d_ws, size_t ws_size,
                              hipStream_t stream) {
  (void)in_sizes; (void)n_in; (void)out_size; (void)ws_size;
  const float* x      = (const float*)d_in[0];
  const float* qkv_w  = (const float*)d_in[1];
  const float* q_bias = (const float*)d_in[2];
  const float* v_bias = (const float*)d_in[3];
  const float* flex   = (const float*)d_in[4];
  const float* cpb_w1 = (const float*)d_in[5];
  const float* cpb_b1 = (const float*)d_in[6];
  const float* cpb_w2 = (const float*)d_in[7];
  const float* proj_w = (const float*)d_in[8];
  const float* proj_b = (const float*)d_in[9];
  float* out = (float*)d_out;

  char* ws = (char*)d_ws;
  float*  rpb      = (float*)(ws + 0);         // 131072 B
  __bf16* qkvw_bf  = (__bf16*)(ws + 131072);   // 393216 B
  __bf16* projw_bf = (__bf16*)(ws + 524288);   // 131072 B
  float*  table    = (float*)(ws + 655360);    // 7200 B

  cpb_table_kernel<<<225, 256, 0, stream>>>(cpb_w1, cpb_b1, cpb_w2, table);
  prep_convert_kernel<<<768, 256, 0, stream>>>(qkv_w, proj_w, table, qkvw_bf, projw_bf, rpb);
  winattn_kernel<<<NWIN, 256, SMEM_BYTES, stream>>>(x, qkvw_bf, q_bias, v_bias, flex,
                                                    rpb, projw_bf, proj_b, out);
}
